// Joint_14525579395592
// MI455X (gfx1250) — matrix operand order repaired and checked by reading, not compiled, not run on hardware
//
#include <hip/hip_runtime.h>

// ---------------------------------------------------------------------------
// RNN-T joint network, fused for MI455X (gfx1250, wave32, WMMA bf16).
//   h = ReLU(f @ W1t^T + g @ W1p^T + (b1t + b1p))    [32768, 512]
//   y = h @ W2^T + b2                                 [32768, 29]
// One fused kernel: h lives only in LDS (bf16), saving ~134MB of HBM traffic.
// Math path: v_wmma_f32_16x16x32_bf16 with fp32 accumulate.
// Roofline: 46 GFLOP vs 180 MB minimum traffic -> sits at the bf16 ridge, so
// WMMA-bf16 + full fusion is the right operating point.
// ---------------------------------------------------------------------------

typedef __attribute__((ext_vector_type(16))) __bf16 v16bf;
typedef __attribute__((ext_vector_type(8)))  __bf16 v8bf;
typedef __attribute__((ext_vector_type(8)))  float  v8f;

#define TRANS_H    1024
#define PRED_H     320
#define KTOT       1344          // TRANS_H + PRED_H (concatenated K)
#define JOINT_H    512
#define NUM_LABELS 29
#define BATCH      32768

#define M_TILE     128           // rows per workgroup
#define THREADS    512           // 16 waves of 32
#define KSTEP      32            // bf16 WMMA K
#define NITER      (KTOT / KSTEP) // 42

// LDS strides in bf16 elements; +8 pad keeps 16B alignment and skews banks.
#define SA_STRIDE  40            // 32 + 8  (80 bytes/row)
#define SH_STRIDE  520           // 512 + 8 (1040 bytes/row)

union Frag16 {                   // one WMMA 16-bit operand (8 VGPRs)
  v16bf v;
  uint4 u[2];
};

__device__ __forceinline__ unsigned short f32_to_bf16(float f) {
  unsigned int u = __float_as_uint(f);
  u += 0x7FFFu + ((u >> 16) & 1u);         // round-to-nearest-even
  return (unsigned short)(u >> 16);
}

// ---------------------------------------------------------------------------
// Prep: W1cat = bf16([W1t | W1p]) [512][1344], W2p = bf16(pad(W2)) [32][512],
// bias1 = b1t + b1p.  ~1.41 MB of workspace, L2-resident afterwards.
// ---------------------------------------------------------------------------
__global__ void prep_kernel(const float* __restrict__ W1t,
                            const float* __restrict__ W1p,
                            const float* __restrict__ b1t,
                            const float* __restrict__ b1p,
                            const float* __restrict__ W2,
                            unsigned short* __restrict__ w1cat,
                            unsigned short* __restrict__ w2p,
                            float* __restrict__ bias1) {
  const int idx = blockIdx.x * blockDim.x + threadIdx.x;
  const int NW1 = JOINT_H * KTOT;   // 688128
  const int NW2 = 32 * JOINT_H;     // 16384
  if (idx < NW1) {
    const int n = idx / KTOT, k = idx - n * KTOT;
    const float v = (k < TRANS_H) ? W1t[n * TRANS_H + k]
                                  : W1p[n * PRED_H + (k - TRANS_H)];
    w1cat[idx] = f32_to_bf16(v);
  } else if (idx < NW1 + NW2) {
    const int i = idx - NW1;
    const int n = i >> 9, k = i & 511;
    const float v = (n < NUM_LABELS) ? W2[n * JOINT_H + k] : 0.0f;
    w2p[i] = f32_to_bf16(v);
  } else if (idx < NW1 + NW2 + JOINT_H) {
    const int i = idx - NW1 - NW2;
    bias1[i] = b1t[i] + b1p[i];
  }
}

// ---------------------------------------------------------------------------
// Activation staging, split so global loads issue at the top of an iteration
// and the LDS writes (which force a load-wait) land AFTER the WMMA block.
// ---------------------------------------------------------------------------
struct StageRegs { float4 a, b; };

__device__ __forceinline__ StageRegs stage_load(const float* __restrict__ f,
                                                const float* __restrict__ g,
                                                int m0, int srow, int skk,
                                                int k0) {
  const float* src;  int stride, kk;
  if (k0 < TRANS_H) { src = f; stride = TRANS_H; kk = k0; }
  else              { src = g; stride = PRED_H;  kk = k0 - TRANS_H; }
  const float* p = src + (size_t)(m0 + srow) * stride + kk + skk;
  StageRegs r;
  r.a = *(const float4*)p;
  r.b = *(const float4*)(p + 4);
  return r;
}

__device__ __forceinline__ void stage_store(unsigned short* __restrict__ dst,
                                            StageRegs r) {
  v8f x;
  x[0] = r.a.x; x[1] = r.a.y; x[2] = r.a.z; x[3] = r.a.w;
  x[4] = r.b.x; x[5] = r.b.y; x[6] = r.b.z; x[7] = r.b.w;
  union { v8bf v; uint4 u; } t;
  t.v = __builtin_convertvector(x, v8bf);   // packed v_cvt_pk_bf16_f32 (RNE)
  *(uint4*)dst = t.u;
}

// ---------------------------------------------------------------------------
// Fused joint kernel.  Grid: BATCH/M_TILE = 256 workgroups of 512 threads.
// ---------------------------------------------------------------------------
__global__ __launch_bounds__(THREADS)
void joint_fused_kernel(const float* __restrict__ f,
                        const float* __restrict__ g,
                        const unsigned short* __restrict__ w1cat,
                        const unsigned short* __restrict__ w2p,
                        const float* __restrict__ bias1,
                        const float* __restrict__ b2,
                        float* __restrict__ out) {
  __shared__ unsigned short sA[2][M_TILE * SA_STRIDE];  // 20 KB (double buf)
  __shared__ unsigned short sH[M_TILE * SH_STRIDE];     // 130 KB (h in bf16)

  const int tid  = threadIdx.x;
  const int lane = tid & 31;
  const int wave = tid >> 5;
  const int m0   = blockIdx.x * M_TILE;

  const int srow = tid >> 2;        // 0..127
  const int skk  = (tid & 3) * 8;   // 0,8,16,24

  // ---- Phase 1: h tile = ReLU(act @ W1cat^T + bias1), kept in LDS --------
  const int arow  = lane & 15;      // row (A) / col (B,C) within 16
  const int ahalf = lane >> 4;      // K-half selector per ISA layouts
  const int n0    = wave * 32;      // each wave owns 32 of JOINT_H's 512 cols

  v8f acc[8][2];
#pragma unroll
  for (int mt = 0; mt < 8; ++mt)
#pragma unroll
    for (int nt = 0; nt < 2; ++nt)
      acc[mt][nt] = (v8f){};

  stage_store(&sA[0][srow * SA_STRIDE + skk],
              stage_load(f, g, m0, srow, skk, 0));
  __syncthreads();

  for (int it = 0; it < NITER; ++it) {
    const int cur = it & 1;
    const int k0  = it * KSTEP;

    // B frags for this K slice: K-contiguous straight from L2-resident bf16.
    Frag16 bfr[2];
#pragma unroll
    for (int nt = 0; nt < 2; ++nt) {
      const int n = n0 + nt * 16 + arow;
      const unsigned short* bp = w1cat + (size_t)n * KTOT + k0 + ahalf * 8;
      bfr[nt].u[0] = *(const uint4*)bp;
      bfr[nt].u[1] = *(const uint4*)(bp + 16);
    }

    // Issue next activation tile's global loads now; store to LDS after WMMA.
    const bool have_next = (it + 1 < NITER);
    StageRegs sr = {};
    if (have_next) sr = stage_load(f, g, m0, srow, skk, (it + 1) * KSTEP);

#pragma unroll
    for (int mt = 0; mt < 8; ++mt) {
      Frag16 afr;                        // A frag: two ds_load_b128
      const unsigned short* ap =
          &sA[cur][(mt * 16 + arow) * SA_STRIDE + ahalf * 8];
      afr.u[0] = *(const uint4*)ap;
      afr.u[1] = *(const uint4*)(ap + 16);
#pragma unroll
      for (int nt = 0; nt < 2; ++nt) {
        acc[mt][nt] = __builtin_amdgcn_wmma_f32_16x16x32_bf16(
            false, afr.v, false, bfr[nt].v, (short)0, acc[mt][nt],
            false, false);
      }
    }

    if (have_next)
      stage_store(&sA[cur ^ 1][srow * SA_STRIDE + skk], sr);
    __syncthreads();
  }

  // bias + ReLU epilogue -> h (bf16) in LDS
  float bv[2];
#pragma unroll
  for (int nt = 0; nt < 2; ++nt) bv[nt] = bias1[n0 + nt * 16 + arow];

#pragma unroll
  for (int mt = 0; mt < 8; ++mt)
#pragma unroll
    for (int nt = 0; nt < 2; ++nt) {
      const int col = n0 + nt * 16 + arow;
#pragma unroll
      for (int r = 0; r < 8; ++r) {      // C layout: vgpr r, half-sel rows
        float v = acc[mt][nt][r] + bv[nt];
        v = v > 0.0f ? v : 0.0f;
        const int row = mt * 16 + r + ahalf * 8;
        sH[row * SH_STRIDE + col] = f32_to_bf16(v);
      }
    }
  __syncthreads();

  // ---- Phase 2: y = h @ W2p^T + b2; 16 waves = 8 Mtiles x 2 Ntiles -------
  const int mt2 = wave >> 1;
  const int nt2 = wave & 1;
  v8f acc2 = (v8f){};
#pragma unroll 4
  for (int j0 = 0; j0 < JOINT_H; j0 += KSTEP) {
    Frag16 afr, bfr;
    const unsigned short* ap =
        &sH[(mt2 * 16 + arow) * SH_STRIDE + j0 + ahalf * 8];
    afr.u[0] = *(const uint4*)ap;
    afr.u[1] = *(const uint4*)(ap + 16);
    const int n = nt2 * 16 + arow;
    const unsigned short* bp = w2p + (size_t)n * JOINT_H + j0 + ahalf * 8;
    bfr.u[0] = *(const uint4*)bp;
    bfr.u[1] = *(const uint4*)(bp + 16);
    acc2 = __builtin_amdgcn_wmma_f32_16x16x32_bf16(
        false, afr.v, false, bfr.v, (short)0, acc2, false, false);
  }

  const int ncol = nt2 * 16 + arow;
  if (ncol < NUM_LABELS) {
    const float bb = b2[ncol];
#pragma unroll
    for (int r = 0; r < 8; ++r) {
      const int row = m0 + mt2 * 16 + r + ahalf * 8;
      out[(size_t)row * NUM_LABELS + ncol] = acc2[r] + bb;
    }
  }
}

// ---------------------------------------------------------------------------
extern "C" void kernel_launch(void* const* d_in, const int* in_sizes, int n_in,
                              void* d_out, int out_size, void* d_ws,
                              size_t ws_size, hipStream_t stream) {
  const float* f   = (const float*)d_in[0];
  const float* g   = (const float*)d_in[1];
  const float* W1t = (const float*)d_in[2];
  const float* b1t = (const float*)d_in[3];
  const float* W1p = (const float*)d_in[4];
  const float* b1p = (const float*)d_in[5];
  const float* W2  = (const float*)d_in[6];
  const float* b2  = (const float*)d_in[7];
  float* out = (float*)d_out;

  // workspace layout (all 16B aligned)
  char* ws = (char*)d_ws;
  unsigned short* w1cat = (unsigned short*)ws;                       // 1376256 B
  unsigned short* w2p =
      (unsigned short*)(ws + (size_t)JOINT_H * KTOT * 2);            //   32768 B
  float* bias1 =
      (float*)(ws + (size_t)JOINT_H * KTOT * 2 + 32 * JOINT_H * 2);  //    2048 B

  const int prepN = JOINT_H * KTOT + 32 * JOINT_H + JOINT_H;
  prep_kernel<<<(prepN + 255) / 256, 256, 0, stream>>>(
      W1t, W1p, b1t, b1p, W2, w1cat, w2p, bias1);

  joint_fused_kernel<<<BATCH / M_TILE, THREADS, 0, stream>>>(
      f, g, w1cat, w2p, bias1, b2, out);
}
